// PointTransformerBlock_54254026883692
// MI455X (gfx1250) — matrix operand order repaired and checked by reading, not compiled, not run on hardware
//
#include <hip/hip_runtime.h>
#include <hip/hip_bf16.h>

typedef __bf16 bf16_t;
typedef bf16_t v16bf __attribute__((ext_vector_type(16)));
typedef bf16_t v8bf  __attribute__((ext_vector_type(8)));
typedef float  v8f   __attribute__((ext_vector_type(8)));
typedef float  v4f   __attribute__((ext_vector_type(4)));

namespace {
constexpr int B_   = 2;
constexpr int N_   = 8192;
constexpr int KNB  = 16;
constexpr int CIN  = 64;
constexpr int HIDC = 256;
constexpr int DM   = 256;      // HEADS * DKQV
constexpr int MTOT = B_ * N_;  // 16384
}

// ---------------------------------------------------------------------------
// Weight prep: fp32 -> bf16 (and fuse Wq/Wk/Wv into one [256 x 768] matrix)
// ---------------------------------------------------------------------------
__global__ void cvt_f32_bf16(const float* __restrict__ src,
                             bf16_t* __restrict__ dst, int n) {
  int i = blockIdx.x * blockDim.x + threadIdx.x;
  if (i < n) dst[i] = (bf16_t)src[i];
}

__global__ void pack_qkv_w(const float* __restrict__ Wq,
                           const float* __restrict__ Wk,
                           const float* __restrict__ Wv,
                           bf16_t* __restrict__ dst) {
  // dst: [2][256][768]; cols 0-255 = Wq, 256-511 = Wk, 512-767 = Wv
  int t = blockIdx.x * blockDim.x + threadIdx.x;
  if (t >= 2 * HIDC * 768) return;
  int l = t / (HIDC * 768);
  int r = (t / 768) % HIDC;
  int c = t % 768;
  float v;
  if (c < 256)      v = Wq[(size_t)l * HIDC * DM + r * DM + c];
  else if (c < 512) v = Wk[(size_t)l * HIDC * DM + r * DM + (c - 256)];
  else              v = Wv[(size_t)l * HIDC * DM + r * DM + (c - 512)];
  dst[t] = (bf16_t)v;
}

__global__ void pack_qkv_b(const float* __restrict__ bq,
                           const float* __restrict__ bk,
                           const float* __restrict__ bv,
                           float* __restrict__ dst) {
  int t = blockIdx.x * blockDim.x + threadIdx.x;
  if (t >= 2 * 768) return;
  int l = t / 768, c = t % 768;
  dst[t] = (c < 256) ? bq[l * DM + c]
         : (c < 512) ? bk[l * DM + (c - 256)]
                     : bv[l * DM + (c - 512)];
}

// ---------------------------------------------------------------------------
// WMMA bf16 GEMM: Out[M,NCOLS] = act(A[M,KDIM] @ W[KDIM,NCOLS] + bias [+res])
// Wave tile: 16 rows x 128 cols (8 accumulators reuse one A fragment).
// Block: 256 threads = 8 waves = 4 m-tiles x 2 n-groups = 64 x 256.
// grid = (MTOT/64, NCOLS/256). Compile-time KDIM/NCOLS: K loop fully
// unrolls and all addresses fold into 24-bit immediate offsets.
// Per K-step: ALL 8 B fragments are staged into distinct registers before
// any WMMA issues, so the 128 16-bit B loads clause together and the 8 WMMAs
// drain with descending partial loadcnt waits instead of serializing.
// No divergence anywhere: EXEC stays all-1s for WMMA.
// ---------------------------------------------------------------------------
template <int KDIM, int NCOLS, bool A_BF16, bool ADD_RES, bool LEAKY,
          bool OUT_BF16>
__global__ __launch_bounds__(256) void gemm_wmma(
    const void* __restrict__ Av, const bf16_t* __restrict__ W,
    const float* __restrict__ bias, const float* __restrict__ res,
    void* __restrict__ Out) {
  constexpr int NSUB = 8;  // 8 n-subtiles of 16 -> 16x128 per wave
  const int lane  = threadIdx.x & 31;
  const int wave  = threadIdx.x >> 5;
  const int mtile = (blockIdx.x * 4 + (wave >> 1)) * 16;
  const int ngrp  = (blockIdx.y * 2 + (wave & 1)) * (16 * NSUB);
  const int mrow  = mtile + (lane & 15);  // A row for this lane
  const int kbase = (lane < 16) ? 0 : 8;  // A-fragment K split

  v8f c[NSUB];
#pragma unroll
  for (int t = 0; t < NSUB; ++t) {
    float bv = bias[ngrp + t * 16 + (lane & 15)];
#pragma unroll
    for (int r = 0; r < 8; ++r) c[t][r] = bv;  // bias folded into accumulator
  }

#pragma unroll
  for (int kb = 0; kb < KDIM; kb += 32) {
    // ---- A fragment (ISA 16-bit A 16x32 layout) ----
    v16bf a;
    if (A_BF16) {
      const bf16_t* arow = (const bf16_t*)Av + (size_t)mrow * KDIM + kb;
      v8bf lo = *(const v8bf*)(arow + kbase);
      v8bf hi = *(const v8bf*)(arow + 16 + kbase);
#pragma unroll
      for (int i = 0; i < 8; ++i) { a[i] = lo[i]; a[8 + i] = hi[i]; }
    } else {
      const float* arow = (const float*)Av + (size_t)mrow * KDIM + kb;
      v4f f0 = *(const v4f*)(arow + kbase);
      v4f f1 = *(const v4f*)(arow + kbase + 4);
      v4f f2 = *(const v4f*)(arow + 16 + kbase);
      v4f f3 = *(const v4f*)(arow + 16 + kbase + 4);
#pragma unroll
      for (int i = 0; i < 4; ++i) {
        a[i]      = (bf16_t)f0[i];
        a[4 + i]  = (bf16_t)f1[i];
        a[8 + i]  = (bf16_t)f2[i];
        a[12 + i] = (bf16_t)f3[i];
      }
    }
    // ---- stage ALL B fragments first (lane = N col, elems = K in A's table order) ----
    const bf16_t* brow = W + (size_t)(kb + kbase) * NCOLS + ngrp + (lane & 15);
    v16bf bfrag[NSUB];
#pragma unroll
    for (int t = 0; t < NSUB; ++t) {
#pragma unroll
      for (int i = 0; i < 8; ++i) { bfrag[t][i] = brow[(size_t)i * NCOLS + t * 16]; bfrag[t][8 + i] = brow[(size_t)(16 + i) * NCOLS + t * 16]; }
    }
    // ---- then issue the 8 WMMAs ----
#pragma unroll
    for (int t = 0; t < NSUB; ++t) {
      c[t] = __builtin_amdgcn_wmma_f32_16x16x32_bf16(
          false, a, false, bfrag[t], (short)0, c[t], false, false);
    }
  }

  // ---- epilogue: lane owns column n, 8 rows; all offsets are immediates ----
  const int mbase    = mtile + ((lane < 16) ? 0 : 8);
  const size_t ebase = (size_t)mbase * NCOLS + ngrp + (lane & 15);
  const float*  rp = ADD_RES ? res + ebase : nullptr;
  bf16_t* ob = (bf16_t*)Out + ebase;
  float*  of = (float*)Out + ebase;
#pragma unroll
  for (int t = 0; t < NSUB; ++t) {
#pragma unroll
    for (int r = 0; r < 8; ++r) {
      float v = c[t][r];
      if (ADD_RES) v += rp[r * NCOLS + t * 16];
      if (LEAKY)   v = (v > 0.f) ? v : 0.01f * v;
      if (OUT_BF16) ob[r * NCOLS + t * 16] = (bf16_t)v;
      else          of[r * NCOLS + t * 16] = v;
    }
  }
}

// ---------------------------------------------------------------------------
// KNN attention: one wave per point; lane owns 8 channels (head = lane/4).
// qkv: [MTOT,768] bf16 (q|k|v). Two gather passes (logits, then V-accum).
// ---------------------------------------------------------------------------
__global__ __launch_bounds__(256) void knn_attn_kernel(
    const bf16_t* __restrict__ qkv, const float* __restrict__ xyz,
    const int* __restrict__ kg, const float* __restrict__ Wp,
    const float* __restrict__ bp, bf16_t* __restrict__ o) {
  const int lane = threadIdx.x & 31;
  const int wave = threadIdx.x >> 5;
  const int p    = blockIdx.x * 8 + wave;  // global point id
  const int b    = p / N_;
  const int d0   = lane * 8;               // this lane's 8 channels

  float qv[8], wp0[8], wp1[8], wp2[8], bpv[8];
  {
    v8bf qb = *(const v8bf*)(qkv + (size_t)p * 768 + d0);
    const float rsc = 0.17677669529663687f;  // 1/sqrt(32)
#pragma unroll
    for (int i = 0; i < 8; ++i) qv[i] = (float)qb[i] * rsc;
#pragma unroll
    for (int i = 0; i < 8; ++i) {
      wp0[i] = Wp[0 * DM + d0 + i];
      wp1[i] = Wp[1 * DM + d0 + i];
      wp2[i] = Wp[2 * DM + d0 + i];
      bpv[i] = bp[d0 + i];
    }
  }
  const float px = xyz[(size_t)p * 3 + 0];
  const float py = xyz[(size_t)p * 3 + 1];
  const float pz = xyz[(size_t)p * 3 + 2];

  // Pass 1: logits over the 16 neighbors (reduce 4 lanes -> one head)
  float logit[KNB];
#pragma unroll
  for (int j = 0; j < KNB; ++j) {
    const int idx = kg[(size_t)p * KNB + j];
    const int pj  = b * N_ + idx;
    const float rx = px - xyz[(size_t)pj * 3 + 0];
    const float ry = py - xyz[(size_t)pj * 3 + 1];
    const float rz = pz - xyz[(size_t)pj * 3 + 2];
    v8bf kv = *(const v8bf*)(qkv + (size_t)pj * 768 + 256 + d0);
    float s = 0.f;
#pragma unroll
    for (int i = 0; i < 8; ++i) {
      float pos = rx * wp0[i] + ry * wp1[i] + rz * wp2[i] + bpv[i];
      s += qv[i] * ((float)kv[i] + pos);
    }
    s += __shfl_xor(s, 1, 32);
    s += __shfl_xor(s, 2, 32);  // now head-wide dot product (32 dims)
    logit[j] = s;
  }

  // register softmax over K=16
  float mx = logit[0];
#pragma unroll
  for (int j = 1; j < KNB; ++j) mx = fmaxf(mx, logit[j]);
  float attn[KNB];
  float ssum = 0.f;
#pragma unroll
  for (int j = 0; j < KNB; ++j) { attn[j] = __expf(logit[j] - mx); ssum += attn[j]; }
  const float inv = 1.f / ssum;
#pragma unroll
  for (int j = 0; j < KNB; ++j) attn[j] *= inv;

  // Pass 2: weighted sum of (v_nb + pos)
  float acc[8];
#pragma unroll
  for (int i = 0; i < 8; ++i) acc[i] = 0.f;
#pragma unroll
  for (int j = 0; j < KNB; ++j) {
    const int idx = kg[(size_t)p * KNB + j];
    const int pj  = b * N_ + idx;
    const float rx = px - xyz[(size_t)pj * 3 + 0];
    const float ry = py - xyz[(size_t)pj * 3 + 1];
    const float rz = pz - xyz[(size_t)pj * 3 + 2];
    v8bf vv = *(const v8bf*)(qkv + (size_t)pj * 768 + 512 + d0);
    const float aw = attn[j];
#pragma unroll
    for (int i = 0; i < 8; ++i) {
      float pos = rx * wp0[i] + ry * wp1[i] + rz * wp2[i] + bpv[i];
      acc[i] += aw * ((float)vv[i] + pos);
    }
  }
  v8bf ob;
#pragma unroll
  for (int i = 0; i < 8; ++i) ob[i] = (bf16_t)acc[i];
  *(v8bf*)(o + (size_t)p * DM + d0) = ob;
}

// ---------------------------------------------------------------------------
extern "C" void kernel_launch(void* const* d_in, const int* in_sizes, int n_in,
                              void* d_out, int out_size, void* d_ws,
                              size_t ws_size, hipStream_t stream) {
  (void)in_sizes; (void)n_in; (void)out_size; (void)ws_size;
  const float* xyz   = (const float*)d_in[0];
  const float* feat  = (const float*)d_in[1];
  const int*   kg    = (const int*)d_in[2];
  const float* W_in  = (const float*)d_in[3];
  const float* b_in  = (const float*)d_in[4];
  const float* Wq    = (const float*)d_in[5];
  const float* bq    = (const float*)d_in[6];
  const float* Wk    = (const float*)d_in[7];
  const float* bk    = (const float*)d_in[8];
  const float* Wv    = (const float*)d_in[9];
  const float* bv    = (const float*)d_in[10];
  const float* Wp    = (const float*)d_in[11];
  const float* bp    = (const float*)d_in[12];
  const float* Wo    = (const float*)d_in[13];
  const float* bo    = (const float*)d_in[14];
  const float* W_out = (const float*)d_in[15];
  const float* b_out = (const float*)d_in[16];
  float* out = (float*)d_out;

  // workspace carve-out (all 256B aligned), total ~51.5 MB
  char* ws = (char*)d_ws;
  size_t off = 0;
  auto take = [&](size_t bytes) -> char* {
    char* pp = ws + off;
    off = (off + bytes + 255) & ~(size_t)255;
    return pp;
  };
  bf16_t* wIn  = (bf16_t*)take((size_t)CIN * DM * 2);
  bf16_t* wQKV = (bf16_t*)take((size_t)2 * HIDC * 768 * 2);
  bf16_t* wO   = (bf16_t*)take((size_t)2 * DM * HIDC * 2);
  bf16_t* wOut = (bf16_t*)take((size_t)HIDC * DM * 2);
  float*  bQKV = (float*) take((size_t)2 * 768 * 4);
  float*  fbuf = (float*) take((size_t)MTOT * HIDC * 4);
  bf16_t* qkv  = (bf16_t*)take((size_t)MTOT * 768 * 2);
  bf16_t* obuf = (bf16_t*)take((size_t)MTOT * DM * 2);

  // weight prep (tiny)
  cvt_f32_bf16<<<dim3((CIN * DM + 255) / 256), dim3(256), 0, stream>>>(
      W_in, wIn, CIN * DM);
  pack_qkv_w<<<dim3((2 * HIDC * 768 + 255) / 256), dim3(256), 0, stream>>>(
      Wq, Wk, Wv, wQKV);
  cvt_f32_bf16<<<dim3((2 * DM * HIDC + 255) / 256), dim3(256), 0, stream>>>(
      Wo, wO, 2 * DM * HIDC);
  cvt_f32_bf16<<<dim3((HIDC * DM + 255) / 256), dim3(256), 0, stream>>>(
      W_out, wOut, HIDC * DM);
  pack_qkv_b<<<dim3(6), dim3(256), 0, stream>>>(bq, bk, bv, bQKV);

  const dim3 blk(256);
  // f = features @ W_in + b_in     (M=16384, K=64, N=256)
  gemm_wmma<CIN, DM, false, false, false, false>
      <<<dim3(MTOT / 64, 1), blk, 0, stream>>>(feat, wIn, b_in, nullptr, fbuf);
  for (int l = 0; l < 2; ++l) {
    // qkv = f @ [Wq|Wk|Wv] + [bq|bk|bv]   (M=16384, K=256, N=768, bf16 out)
    gemm_wmma<HIDC, 768, false, false, false, true>
        <<<dim3(MTOT / 64, 3), blk, 0, stream>>>(
            fbuf, wQKV + (size_t)l * HIDC * 768, bQKV + l * 768, nullptr, qkv);
    // o = knn-attention(q,k,v,pos)
    knn_attn_kernel<<<dim3(MTOT / 8), blk, 0, stream>>>(
        qkv, xyz, kg, Wp + (size_t)l * 3 * DM, bp + (size_t)l * DM, obuf);
    // f = o @ Wo + bo + f   (residual, in-place, bf16 A operand)
    gemm_wmma<DM, HIDC, true, true, false, false>
        <<<dim3(MTOT / 64, 1), blk, 0, stream>>>(
            obuf, wO + (size_t)l * DM * HIDC, bo + (size_t)l * HIDC, fbuf,
            fbuf);
  }
  // out = leaky_relu(f @ W_out + b_out)
  gemm_wmma<HIDC, DM, false, false, true, false>
      <<<dim3(MTOT / 64, 1), blk, 0, stream>>>(fbuf, wOut, b_out, nullptr, out);
}
